// ScaledAttentionLayer_11029476016303
// MI455X (gfx1250) — compile-verified, matrix operand order checked by reading, not run on hardware
//
#include <hip/hip_runtime.h>
#include <hip/hip_bf16.h>

#define BB 8
#define NN 2048
#define DD 512
#define SCALE 0.044194173824159216f  // 1/sqrt(512)

typedef __attribute__((ext_vector_type(16))) __bf16 v16bf;
typedef __attribute__((ext_vector_type(8)))  float  v8f;

// ---- CDNA5 WMMA helpers (CDNA5 ISA layouts, wave32) -------------------

static __device__ __forceinline__ int kpat(int j, int half) {
  // 16-bit A-matrix 16x32: lanes 0-15 K = {0..7,16..23}, lanes 16-31 K = {8..15,24..31}
  return ((j < 4) ? (2 * j) : (16 + 2 * (j - 4))) + (half ? 8 : 0);
}

// A (or B for C = X * W^T) fragment from a row-major bf16 matrix: element (row, k)
static __device__ __forceinline__ v16bf frag_row_bf16(const __bf16* base, int ld) {
  const int lane = threadIdx.x & 31;
  const int half = lane >> 4;
  const __bf16* p = base + (size_t)(lane & 15) * ld;
  v16bf f;
#pragma unroll
  for (int j = 0; j < 8; ++j) {
    const int k = kpat(j, half);
    f[2 * j]     = p[k];
    f[2 * j + 1] = p[k + 1];
  }
  return f;
}

// Same, but source is f32 (cast to bf16 on the fly)
static __device__ __forceinline__ v16bf frag_row_f32(const float* base, int ld) {
  const int lane = threadIdx.x & 31;
  const int half = lane >> 4;
  const float* p = base + (size_t)(lane & 15) * ld;
  v16bf f;
#pragma unroll
  for (int j = 0; j < 8; ++j) {
    const int k = kpat(j, half);
    f[2 * j]     = (__bf16)p[k];
    f[2 * j + 1] = (__bf16)p[k + 1];
  }
  return f;
}

// B fragment from a row-major matrix accessed column-wise: element (k, col)
static __device__ __forceinline__ v16bf frag_col_bf16(const __bf16* base, int ld) {
  const int lane = threadIdx.x & 31;
  const int half = lane >> 4;
  const __bf16* p = base + (lane & 15);
  v16bf f;
#pragma unroll
  for (int j = 0; j < 8; ++j) {
    const int k = kpat(j, half);
    f[2 * j]     = p[(size_t)k * ld];
    f[2 * j + 1] = p[(size_t)(k + 1) * ld];
  }
  return f;
}

static __device__ __forceinline__ v8f wmma_bf16(v16bf a, v16bf b, v8f c) {
  return __builtin_amdgcn_wmma_f32_16x16x32_bf16(false, a, false, b, (short)0, c,
                                                 false, false);
}

// ---- Kernel 1: q = (x*Wq^T + bq)*scale, v = ctx*Wv^T + bv, k = bf16(ctx) --
__global__ void __launch_bounds__(256)
proj_qv_kernel(const float* __restrict__ query, const float* __restrict__ context,
               const float* __restrict__ Wq, const float* __restrict__ bq,
               const float* __restrict__ Wv, const float* __restrict__ bv,
               __bf16* __restrict__ qs, __bf16* __restrict__ kb,
               __bf16* __restrict__ vb) {
  const int wave = threadIdx.x >> 5;
  const int tile = blockIdx.x * 8 + wave;          // 32768 tiles total
  const int mt = tile >> 5;                        // row tile  (/ (DD/16))
  const int nt = tile & 31;                        // col tile  (% (DD/16))

  const float* xq = query   + (size_t)mt * 16 * DD;
  const float* xv = context + (size_t)mt * 16 * DD;
  const float* wq = Wq + (size_t)nt * 16 * DD;
  const float* wv = Wv + (size_t)nt * 16 * DD;

  v8f accq = {};
  v8f accv = {};
#pragma unroll 4
  for (int kk = 0; kk < DD; kk += 32) {
    v16bf aq  = frag_row_f32(xq + kk, DD);
    v16bf wqf = frag_row_f32(wq + kk, DD);
    accq = wmma_bf16(aq, wqf, accq);
    v16bf av  = frag_row_f32(xv + kk, DD);
    v16bf wvf = frag_row_f32(wv + kk, DD);
    accv = wmma_bf16(av, wvf, accv);
  }

  const int lane = threadIdx.x & 31;
  const int half = lane >> 4;
  const int col  = lane & 15;
  const float biasq = bq[nt * 16 + col];
  const float biasv = bv[nt * 16 + col];
#pragma unroll
  for (int r = 0; r < 8; ++r) {
    const int m = mt * 16 + r + half * 8;
    qs[(size_t)m * DD + nt * 16 + col] = (__bf16)((accq[r] + biasq) * SCALE);
    vb[(size_t)m * DD + nt * 16 + col] = (__bf16)(accv[r] + biasv);
  }

  const size_t total = (size_t)BB * NN * DD;
  for (size_t i = (size_t)blockIdx.x * blockDim.x + threadIdx.x; i < total;
       i += (size_t)gridDim.x * blockDim.x)
    kb[i] = (__bf16)context[i];
}

// ---- Kernel 2: attention, one (batch, 16-query tile) per 4-wave block -----
// LDS: 16KB Q tile (async-staged) + 128KB f32 scores + 64KB bf16 probs + red.
#define SM_Q   0
#define SM_SC  (16 * DD * 2)                         // 16384
#define SM_PR  (SM_SC + 16 * NN * 4)                 // 147456
#define SM_RED (SM_PR + 16 * NN * 2)                 // 212992
#define SM_RMX (SM_RED + 16 * 8 * 4)                 // 213504
#define SM_RIV (SM_RMX + 16 * 4)                     // 213568
#define SM_TOT (SM_RIV + 16 * 4)                     // 213632 bytes (< 320KB/WGP)

__global__ void __launch_bounds__(128)
attn_kernel(const __bf16* __restrict__ qs, const __bf16* __restrict__ kb,
            const __bf16* __restrict__ vb, __bf16* __restrict__ hb) {
  extern __shared__ char smem[];
  __bf16* qtile  = (__bf16*)(smem + SM_Q);    // [16][512]
  float*  sc     = (float*)(smem + SM_SC);    // [16][2048]
  __bf16* pr     = (__bf16*)(smem + SM_PR);   // [16][2048]
  float*  red    = (float*)(smem + SM_RED);   // [16][8]
  float*  rowmax = (float*)(smem + SM_RMX);   // [16]
  float*  rowinv = (float*)(smem + SM_RIV);   // [16]

  const int b    = blockIdx.x >> 7;           // / (NN/16)
  const int qt   = blockIdx.x & 127;
  const int tid  = threadIdx.x;
  const int wave = tid >> 5;
  const int lane = tid & 31;
  const int half = lane >> 4;
  const int col  = lane & 15;

  // ---- Phase 0: async-stage the 16x512 bf16 Q tile into LDS --------------
  // 16KB total; 128 threads x 8 x B128 (16B) ops. (ISA 15.18.3 op 98)
  {
    const char* gq = (const char*)(qs + ((size_t)b * NN + qt * 16) * DD);
#pragma unroll
    for (int i = 0; i < 8; ++i) {
      const unsigned off   = (unsigned)(i * 2048 + tid * 16);
      const unsigned ldsa  = (unsigned)(size_t)(void*)(smem + SM_Q) + off;
      const void*    gaddr = gq + off;
      asm volatile("global_load_async_to_lds_b128 %0, %1, off"
                   :: "v"(ldsa), "v"(gaddr) : "memory");
    }
    asm volatile("s_wait_asynccnt 0x0" ::: "memory");
  }
  __syncthreads();

  // ---- Phase 1: S = (q*scale) @ K^T -> LDS scores; waves split key tiles -
  for (int kt = wave; kt < NN / 16; kt += 4) {
    const __bf16* kbase = kb + ((size_t)b * NN + kt * 16) * DD;
    if (kt + 4 < NN / 16)
      __builtin_prefetch(kbase + 4 * 16 * DD, 0, 1);  // global_prefetch_b8
    v8f acc = {};
#pragma unroll 4
    for (int kk = 0; kk < DD; kk += 32) {
      v16bf a  = frag_row_bf16(qtile + kk, DD);       // ds_load path
      v16bf bf = frag_row_bf16(kbase + kk, DD);       // B[k=d][n=key] = K[key][d]
      acc = wmma_bf16(a, bf, acc);
    }
    float* outp = sc + kt * 16 + col;
#pragma unroll
    for (int r = 0; r < 8; ++r)
      outp[(size_t)(r + half * 8) * NN] = acc[r];
  }
  __syncthreads();

  // ---- Phase 2: softmax over 2048 cols, 128 threads ----------------------
  // thread -> (row = tid&15, segment = tid>>4 of 256 columns)
  {
    const int row = tid & 15;
    const int seg = tid >> 4;
    const float* r0 = sc + (size_t)row * NN + seg * 256;
    float m = -3.4e38f;
    for (int i = 0; i < 256; ++i) m = fmaxf(m, r0[i]);
    red[row * 8 + seg] = m;
  }
  __syncthreads();
  if (tid < 16) {
    float m = red[tid * 8];
#pragma unroll
    for (int j = 1; j < 8; ++j) m = fmaxf(m, red[tid * 8 + j]);
    rowmax[tid] = m;
  }
  __syncthreads();
  {
    const int row = tid & 15;
    const int seg = tid >> 4;
    const float m = rowmax[row];
    const float* r0 = sc + (size_t)row * NN + seg * 256;
    __bf16* p0 = pr + (size_t)row * NN + seg * 256;
    float s = 0.f;
    for (int i = 0; i < 256; ++i) {
      const float e = __expf(r0[i] - m);
      s += e;
      p0[i] = (__bf16)e;
    }
    red[row * 8 + seg] = s;
  }
  __syncthreads();
  if (tid < 16) {
    float s = red[tid * 8];
#pragma unroll
    for (int j = 1; j < 8; ++j) s += red[tid * 8 + j];
    rowinv[tid] = 1.f / s;
  }
  __syncthreads();

  // ---- Phase 3: H = P @ V; waves split the 32 output-d tiles -------------
  const size_t obase = ((size_t)b * NN + qt * 16) * DD;
  for (int nt = wave; nt < DD / 16; nt += 4) {
    v8f acc = {};
    const __bf16* vbase = vb + (size_t)b * NN * DD + nt * 16;
#pragma unroll 2
    for (int kk = 0; kk < NN; kk += 32) {
      v16bf a  = frag_row_bf16(pr + kk, NN);                 // P tile from LDS
      v16bf bf = frag_col_bf16(vbase + (size_t)kk * DD, DD); // V[k=key][n=d]
      acc = wmma_bf16(a, bf, acc);
    }
#pragma unroll
    for (int r = 0; r < 8; ++r) {
      const int m = r + half * 8;
      hb[obase + (size_t)m * DD + nt * 16 + col] = (__bf16)(acc[r] * rowinv[m]);
    }
  }
}

// ---- Kernel 3: out = h @ Wo^T + bo (f32 out) ------------------------------
__global__ void __launch_bounds__(256)
out_proj_kernel(const __bf16* __restrict__ hb, const float* __restrict__ Wo,
                const float* __restrict__ bo, float* __restrict__ out) {
  const int wave = threadIdx.x >> 5;
  const int tile = blockIdx.x * 8 + wave;
  const int mt = tile >> 5;
  const int nt = tile & 31;

  const __bf16* x = hb + (size_t)mt * 16 * DD;
  const float*  w = Wo + (size_t)nt * 16 * DD;
  v8f acc = {};
#pragma unroll 4
  for (int kk = 0; kk < DD; kk += 32) {
    v16bf a  = frag_row_bf16(x + kk, DD);
    v16bf bf = frag_row_f32(w + kk, DD);
    acc = wmma_bf16(a, bf, acc);
  }
  const int lane = threadIdx.x & 31;
  const int half = lane >> 4;
  const int col  = lane & 15;
  const float bias = bo[nt * 16 + col];
#pragma unroll
  for (int r = 0; r < 8; ++r) {
    const int m = mt * 16 + r + half * 8;
    out[(size_t)m * DD + nt * 16 + col] = acc[r] + bias;
  }
}

extern "C" void kernel_launch(void* const* d_in, const int* in_sizes, int n_in,
                              void* d_out, int out_size, void* d_ws, size_t ws_size,
                              hipStream_t stream) {
  const float* query   = (const float*)d_in[0];
  const float* context = (const float*)d_in[1];
  const float* Wq = (const float*)d_in[2];
  const float* bq = (const float*)d_in[3];
  const float* Wv = (const float*)d_in[4];
  const float* bv = (const float*)d_in[5];
  const float* Wo = (const float*)d_in[6];
  const float* bo = (const float*)d_in[7];
  float* out = (float*)d_out;

  const size_t S = (size_t)BB * NN * DD;           // elements per bf16 tensor
  __bf16* qs = (__bf16*)d_ws;
  __bf16* kb = qs + S;
  __bf16* vb = kb + S;
  __bf16* hb = vb + S;                             // total 64 MiB of d_ws

  // 1) projections: (16384/16)*(512/16) = 32768 tiles, 8 waves/block
  proj_qv_kernel<<<4096, 256, 0, stream>>>(query, context, Wq, bq, Wv, bv, qs, kb, vb);

  // 2) attention: 4-wave block per (b, qtile); ~209KB dynamic LDS (<= 320KB/WGP)
  attn_kernel<<<BB * (NN / 16), 128, SM_TOT, stream>>>(qs, kb, vb, hb);

  // 3) output projection
  out_proj_kernel<<<4096, 256, 0, stream>>>(hb, Wo, bo, out);
}
